// DiTBlock_43843026158180
// MI455X (gfx1250) — matrix operand order repaired and checked by reading, not compiled, not run on hardware
//
#include <hip/hip_runtime.h>
#include <hip/hip_bf16.h>
#include <stdint.h>

// ---------------------------------------------------------------------------
// Types & helpers
// ---------------------------------------------------------------------------
typedef __bf16 bf16x16 __attribute__((ext_vector_type(16)));
typedef float  f32x8   __attribute__((ext_vector_type(8)));

union FragU {
    uint4   q[2];
    bf16x16 v;
};

__device__ __forceinline__ unsigned short f2bf(float f) {
    uint32_t u = __float_as_uint(f);
    uint32_t r = 0x7FFFu + ((u >> 16) & 1u);   // round-to-nearest-even
    return (unsigned short)((u + r) >> 16);
}

// A-operand (16x32, MxK): lane = M (0..15), lane-half selects K {0..15 | 16..31},
// 16 consecutive K per lane.
// p = base + (m0 + lane%16)*ld + k0 + (lane/16)*16
__device__ __forceinline__ bf16x16 load_a(const unsigned short* p) {
    FragU f;
    f.q[0] = *(const uint4*)(p);
    f.q[1] = *(const uint4*)(p + 8);
    return f.v;
}

// B-operand (32x16, KxN): lane = N (0..15), lane-half selects K {0..15 | 16..31},
// 16 consecutive K per lane.  p = base + (n0 + lane%16)*ld + k0 + (lane/16)*16
__device__ __forceinline__ bf16x16 load_b(const unsigned short* p) {
    FragU f;
    f.q[0] = *(const uint4*)(p);
    f.q[1] = *(const uint4*)(p + 8);
    return f.v;
}

__device__ __forceinline__ f32x8 wmma_bf16(bf16x16 a, bf16x16 b, f32x8 c) {
    return __builtin_amdgcn_wmma_f32_16x16x32_bf16(
        /*neg_a=*/false, a, /*neg_b=*/false, b,
        /*c_mod=*/(short)0, c, /*reuse_a=*/false, /*reuse_b=*/false);
}

// Async 16-byte global -> LDS copy (per-lane), tracked by ASYNCcnt.
__device__ __forceinline__ void async_copy_b128(const unsigned short* gptr,
                                                unsigned short* lptr) {
    // Flat shared address low 32 bits == LDS byte offset (aperture mapping).
    uint32_t lds = (uint32_t)(uintptr_t)lptr;
    asm volatile("global_load_async_to_lds_b128 %0, %1, off"
                 :: "v"(lds), "v"(gptr) : "memory");
}
__device__ __forceinline__ void wait_async0() {
    asm volatile("s_wait_asynccnt 0" ::: "memory");
}

// ---------------------------------------------------------------------------
// 1) f32 -> bf16 weight conversion
// ---------------------------------------------------------------------------
__global__ void __launch_bounds__(256) cvt_bf16(const float* __restrict__ src,
                                                unsigned short* __restrict__ dst,
                                                int n) {
    int i = (blockIdx.x * 256 + threadIdx.x) * 4;
    if (i < n) {
        float4 v = *(const float4*)(src + i);
        ushort4 o;
        o.x = f2bf(v.x); o.y = f2bf(v.y); o.z = f2bf(v.z); o.w = f2bf(v.w);
        *(ushort4*)(dst + i) = o;
    }
}

// ---------------------------------------------------------------------------
// 2) per-batch modulation vectors:  out[b, o] = dot(t[b, :256], W[o, :256])
// ---------------------------------------------------------------------------
__global__ void __launch_bounds__(256) tvec(const float* __restrict__ t,
                                            const float* __restrict__ W,
                                            float* __restrict__ out) {
    __shared__ float ts[256];
    int b = blockIdx.x;
    int o = blockIdx.y * 256 + threadIdx.x;
    ts[threadIdx.x] = t[b * 256 + threadIdx.x];
    __syncthreads();
    const float* wr = W + (size_t)o * 256;
    float acc = 0.f;
    #pragma unroll 8
    for (int k = 0; k < 256; ++k) acc += ts[k] * wr[k];
    out[b * 1024 + o] = acc;
}

// ---------------------------------------------------------------------------
// 3) AdaRMSNorm -> bf16.  One wave per token (wave32), 32 f32 per lane.
// ---------------------------------------------------------------------------
__global__ void __launch_bounds__(256) ada_rmsnorm_bf16(
        const float* __restrict__ X, const float* __restrict__ gamma,
        const float* __restrict__ beta, unsigned short* __restrict__ Y) {
    int token = blockIdx.x * 8 + (threadIdx.x >> 5);
    int lane  = threadIdx.x & 31;
    int b     = token >> 11;                       // 2048 tokens per batch
    const float* xr = X + (size_t)token * 1024;

    float4 xv[8];
    float  ss = 0.f;
    #pragma unroll
    for (int i = 0; i < 8; ++i) {
        xv[i] = *(const float4*)(xr + lane * 4 + i * 128);
        ss += xv[i].x * xv[i].x + xv[i].y * xv[i].y
            + xv[i].z * xv[i].z + xv[i].w * xv[i].w;
    }
    #pragma unroll
    for (int d = 1; d < 32; d <<= 1) ss += __shfl_xor(ss, d, 32);
    float rinv = rsqrtf(ss * (1.f / 1024.f) + 1e-6f);

    const float* gr = gamma + (size_t)b * 1024;
    const float* br = beta  + (size_t)b * 1024;
    unsigned short* yr = Y + (size_t)token * 1024;
    #pragma unroll
    for (int i = 0; i < 8; ++i) {
        int c = lane * 4 + i * 128;
        float4 g  = *(const float4*)(gr + c);
        float4 bb = *(const float4*)(br + c);
        ushort4 o;
        o.x = f2bf(g.x * xv[i].x * rinv + bb.x);
        o.y = f2bf(g.y * xv[i].y * rinv + bb.y);
        o.z = f2bf(g.z * xv[i].z * rinv + bb.z);
        o.w = f2bf(g.w * xv[i].w * rinv + bb.w);
        *(ushort4*)(yr + c) = o;
    }
}

// ---------------------------------------------------------------------------
// 4) GEMM: Y(bf16)[M,N] = X(bf16)[M,K] @ W(bf16)[N,K]^T
//    block = 256 (8 waves, 2Mx4N), wave tile 32x64, block tile 64x256
//    grid = (M/64, N/256)
// ---------------------------------------------------------------------------
__global__ void __launch_bounds__(256) gemm_proj(
        const unsigned short* __restrict__ X,
        const unsigned short* __restrict__ W,
        unsigned short* __restrict__ Y, int K, int N) {
    const int lane = threadIdx.x & 31;
    const int wave = threadIdx.x >> 5;
    const int M0 = blockIdx.x * 64  + (wave >> 2) * 32;
    const int N0 = blockIdx.y * 256 + (wave & 3) * 64;
    const int mr = lane & 15, hf = lane >> 4;

    f32x8 c[2][4] = {};
    const unsigned short* pa0 = X + (size_t)(M0 + mr) * K + hf * 16;
    const unsigned short* pa1 = pa0 + (size_t)16 * K;
    const unsigned short* pb[4];
    #pragma unroll
    for (int j = 0; j < 4; ++j)
        pb[j] = W + (size_t)(N0 + j * 16 + mr) * K + hf * 16;

    for (int k0 = 0; k0 < K; k0 += 32) {
        __builtin_prefetch(pa0 + k0 + 128, 0, 1);
        __builtin_prefetch(pb[0] + k0 + 128, 0, 1);
        bf16x16 a0 = load_a(pa0 + k0), a1 = load_a(pa1 + k0);
        #pragma unroll
        for (int j = 0; j < 4; ++j) {
            bf16x16 b = load_b(pb[j] + k0);
            c[0][j] = wmma_bf16(a0, b, c[0][j]);
            c[1][j] = wmma_bf16(a1, b, c[1][j]);
        }
    }
    #pragma unroll
    for (int i = 0; i < 2; ++i)
        #pragma unroll
        for (int j = 0; j < 4; ++j) {
            int n = N0 + j * 16 + mr;
            #pragma unroll
            for (int r = 0; r < 8; ++r) {
                int m = M0 + i * 16 + r + 8 * hf;
                Y[(size_t)m * N + n] = f2bf(c[i][j][r]);
            }
        }
}

// ---------------------------------------------------------------------------
// 5) Flash attention: grid = (16 qblocks, 16 heads, 4 batch), block = 256
//    each wave owns a 16-row Q tile; 32-kv blocks staged in LDS
//    (K tile via async global->LDS DMA, V tile transposed through VGPRs)
// ---------------------------------------------------------------------------
__global__ void __launch_bounds__(256) flash_attn(
        const unsigned short* __restrict__ Q,
        const unsigned short* __restrict__ Kb,
        const unsigned short* __restrict__ V,
        unsigned short* __restrict__ O) {
    const int lane = threadIdx.x & 31;
    const int wave = threadIdx.x >> 5;
    const int head = blockIdx.y, batch = blockIdx.z;
    const int q0 = batch * 2048 + blockIdx.x * 128 + wave * 16;
    const int d0 = head * 64;
    const int mr = lane & 15, hf = lane >> 4;

    __shared__ __align__(16) unsigned short Ks[32][64];
    __shared__ __align__(16) unsigned short Vt[64][32];
    __shared__ __align__(16) unsigned short Ps[8][16][32];

    const unsigned short* qp = Q + (size_t)(q0 + mr) * 1024 + d0 + hf * 16;
    bf16x16 qa0 = load_a(qp);
    bf16x16 qa1 = load_a(qp + 32);

    f32x8 oacc[4] = {};
    float mrow[8], lrow[8];
    #pragma unroll
    for (int r = 0; r < 8; ++r) { mrow[r] = -1e30f; lrow[r] = 0.f; }

    const int ldr = threadIdx.x >> 3;          // 0..31 kv row
    const int ldc = (threadIdx.x & 7) * 8;     // 0..56 d col
    const float scale = 0.125f;                // 1/sqrt(64)

    for (int kv0 = batch * 2048; kv0 < (batch + 1) * 2048; kv0 += 32) {
        __syncthreads();
        // K tile: async DMA straight into LDS (16B per lane)
        async_copy_b128(Kb + (size_t)(kv0 + ldr) * 1024 + d0 + ldc,
                        &Ks[ldr][ldc]);
        // V tile: through VGPRs, stored transposed for contiguous B-frags
        union { uint4 q; unsigned short s[8]; } vq;
        vq.q = *(const uint4*)(V + (size_t)(kv0 + ldr) * 1024 + d0 + ldc);
        #pragma unroll
        for (int e = 0; e < 8; ++e) Vt[ldc + e][ldr] = vq.s[e];
        wait_async0();
        __syncthreads();

        // S = Q K^T : two 16x16 tiles (kv 0..15, 16..31), K-dim = d (2x32)
        f32x8 s0 = {}, s1 = {};
        {
            const unsigned short* kp0 = &Ks[mr][hf * 16];
            const unsigned short* kp1 = &Ks[16 + mr][hf * 16];
            s0 = wmma_bf16(qa0, load_b(kp0),      s0);
            s0 = wmma_bf16(qa1, load_b(kp0 + 32), s0);
            s1 = wmma_bf16(qa0, load_b(kp1),      s1);
            s1 = wmma_bf16(qa1, load_b(kp1 + 32), s1);
        }

        // online softmax; rows of C live in (vgpr r, lane-half) -> reduce width 16
        unsigned short* pp = &Ps[wave][0][0];
        #pragma unroll
        for (int r = 0; r < 8; ++r) {
            float v0 = s0[r] * scale, v1 = s1[r] * scale;
            float mx = fmaxf(v0, v1);
            #pragma unroll
            for (int d = 1; d < 16; d <<= 1) mx = fmaxf(mx, __shfl_xor(mx, d, 16));
            float mnew = fmaxf(mrow[r], mx);
            float corr = __expf(mrow[r] - mnew);
            float p0 = __expf(v0 - mnew);
            float p1 = __expf(v1 - mnew);
            float sm = p0 + p1;
            #pragma unroll
            for (int d = 1; d < 16; d <<= 1) sm += __shfl_xor(sm, d, 16);
            lrow[r] = lrow[r] * corr + sm;
            mrow[r] = mnew;
            #pragma unroll
            for (int t = 0; t < 4; ++t) oacc[t][r] *= corr;
            int m = r + 8 * hf;
            pp[m * 32 + mr]      = f2bf(p0);
            pp[m * 32 + 16 + mr] = f2bf(p1);
        }

        // reload P in A-frag layout (same-wave LDS is in-order)
        bf16x16 pa = load_a(&Ps[wave][mr][hf * 16]);
        #pragma unroll
        for (int t = 0; t < 4; ++t)
            oacc[t] = wmma_bf16(pa, load_b(&Vt[t * 16 + mr][hf * 16]), oacc[t]);
    }

    #pragma unroll
    for (int t = 0; t < 4; ++t) {
        int n = d0 + t * 16 + mr;
        #pragma unroll
        for (int r = 0; r < 8; ++r) {
            int m = q0 + r + 8 * hf;
            O[(size_t)m * 1024 + n] = f2bf(oacc[t][r] / lrow[r]);
        }
    }
}

// ---------------------------------------------------------------------------
// 6) O-projection + residual:  out = x + alpha[b] * (O @ Wo^T)    (f32 out)
//    wave tile 32x64, grid = (M/64, 1024/256)
// ---------------------------------------------------------------------------
__global__ void __launch_bounds__(256) gemm_wo_res(
        const unsigned short* __restrict__ X,
        const unsigned short* __restrict__ W,
        const float* __restrict__ xin, const float* __restrict__ alpha,
        float* __restrict__ out) {
    const int K = 1024, N = 1024;
    const int lane = threadIdx.x & 31;
    const int wave = threadIdx.x >> 5;
    const int M0 = blockIdx.x * 64  + (wave >> 2) * 32;
    const int N0 = blockIdx.y * 256 + (wave & 3) * 64;
    const int mr = lane & 15, hf = lane >> 4;

    f32x8 c[2][4] = {};
    const unsigned short* pa0 = X + (size_t)(M0 + mr) * K + hf * 16;
    const unsigned short* pa1 = pa0 + (size_t)16 * K;
    const unsigned short* pb[4];
    #pragma unroll
    for (int j = 0; j < 4; ++j)
        pb[j] = W + (size_t)(N0 + j * 16 + mr) * K + hf * 16;

    for (int k0 = 0; k0 < K; k0 += 32) {
        __builtin_prefetch(pa0 + k0 + 128, 0, 1);
        __builtin_prefetch(pb[0] + k0 + 128, 0, 1);
        bf16x16 a0 = load_a(pa0 + k0), a1 = load_a(pa1 + k0);
        #pragma unroll
        for (int j = 0; j < 4; ++j) {
            bf16x16 b = load_b(pb[j] + k0);
            c[0][j] = wmma_bf16(a0, b, c[0][j]);
            c[1][j] = wmma_bf16(a1, b, c[1][j]);
        }
    }
    #pragma unroll
    for (int i = 0; i < 2; ++i)
        #pragma unroll
        for (int j = 0; j < 4; ++j) {
            int n = N0 + j * 16 + mr;
            #pragma unroll
            for (int r = 0; r < 8; ++r) {
                int m = M0 + i * 16 + r + 8 * hf;
                size_t idx = (size_t)m * N + n;
                out[idx] = xin[idx] + alpha[(m >> 11) * 1024 + n] * c[i][j][r];
            }
        }
}

// ---------------------------------------------------------------------------
// 7) SwiGLU GEMM: H(bf16)[M,4096] = silu(Xn@Wg^T) * (Xn@Wh^T)
//    wave tile 32x32 for BOTH gate & hidden, grid = (M/64, 4096/128)
// ---------------------------------------------------------------------------
__global__ void __launch_bounds__(256) gemm_swiglu(
        const unsigned short* __restrict__ X,
        const unsigned short* __restrict__ Wg,
        const unsigned short* __restrict__ Wh,
        unsigned short* __restrict__ H) {
    const int K = 1024, N = 4096;
    const int lane = threadIdx.x & 31;
    const int wave = threadIdx.x >> 5;
    const int M0 = blockIdx.x * 64  + (wave >> 2) * 32;
    const int N0 = blockIdx.y * 128 + (wave & 3) * 32;
    const int mr = lane & 15, hf = lane >> 4;

    f32x8 cg[2][2] = {}, ch[2][2] = {};
    const unsigned short* pa0 = X  + (size_t)(M0 + mr) * K + hf * 16;
    const unsigned short* pa1 = pa0 + (size_t)16 * K;
    const unsigned short* pg0 = Wg + (size_t)(N0 + mr) * K + hf * 16;
    const unsigned short* pg1 = pg0 + (size_t)16 * K;
    const unsigned short* ph0 = Wh + (size_t)(N0 + mr) * K + hf * 16;
    const unsigned short* ph1 = ph0 + (size_t)16 * K;
    for (int k0 = 0; k0 < K; k0 += 32) {
        __builtin_prefetch(pa0 + k0 + 128, 0, 1);
        bf16x16 a0 = load_a(pa0 + k0), a1 = load_a(pa1 + k0);
        bf16x16 g0 = load_b(pg0 + k0), g1 = load_b(pg1 + k0);
        bf16x16 h0 = load_b(ph0 + k0), h1 = load_b(ph1 + k0);
        cg[0][0] = wmma_bf16(a0, g0, cg[0][0]);
        cg[0][1] = wmma_bf16(a0, g1, cg[0][1]);
        cg[1][0] = wmma_bf16(a1, g0, cg[1][0]);
        cg[1][1] = wmma_bf16(a1, g1, cg[1][1]);
        ch[0][0] = wmma_bf16(a0, h0, ch[0][0]);
        ch[0][1] = wmma_bf16(a0, h1, ch[0][1]);
        ch[1][0] = wmma_bf16(a1, h0, ch[1][0]);
        ch[1][1] = wmma_bf16(a1, h1, ch[1][1]);
    }
    #pragma unroll
    for (int i = 0; i < 2; ++i)
        #pragma unroll
        for (int j = 0; j < 2; ++j) {
            int n = N0 + j * 16 + mr;
            #pragma unroll
            for (int r = 0; r < 8; ++r) {
                int m = M0 + i * 16 + r + 8 * hf;
                float g  = cg[i][j][r];
                float hv = ch[i][j][r];
                float s  = g / (1.f + __expf(-g));   // silu
                H[(size_t)m * N + n] = f2bf(s * hv);
            }
        }
}

// ---------------------------------------------------------------------------
// 8) FFN out GEMM + bias + residual accumulate into d_out (K = 4096)
//    wave tile 32x64, grid = (M/64, 1024/256)
// ---------------------------------------------------------------------------
__global__ void __launch_bounds__(256) gemm_out_res(
        const unsigned short* __restrict__ X,      // H [M,4096]
        const unsigned short* __restrict__ W,      // out_w [1024,4096]
        const float* __restrict__ bias, const float* __restrict__ alpha,
        float* __restrict__ out) {
    const int K = 4096, N = 1024;
    const int lane = threadIdx.x & 31;
    const int wave = threadIdx.x >> 5;
    const int M0 = blockIdx.x * 64  + (wave >> 2) * 32;
    const int N0 = blockIdx.y * 256 + (wave & 3) * 64;
    const int mr = lane & 15, hf = lane >> 4;

    f32x8 c[2][4] = {};
    const unsigned short* pa0 = X + (size_t)(M0 + mr) * K + hf * 16;
    const unsigned short* pa1 = pa0 + (size_t)16 * K;
    const unsigned short* pb[4];
    #pragma unroll
    for (int j = 0; j < 4; ++j)
        pb[j] = W + (size_t)(N0 + j * 16 + mr) * K + hf * 16;

    for (int k0 = 0; k0 < K; k0 += 32) {
        __builtin_prefetch(pa0 + k0 + 128, 0, 1);
        __builtin_prefetch(pb[0] + k0 + 128, 0, 1);
        bf16x16 a0 = load_a(pa0 + k0), a1 = load_a(pa1 + k0);
        #pragma unroll
        for (int j = 0; j < 4; ++j) {
            bf16x16 b = load_b(pb[j] + k0);
            c[0][j] = wmma_bf16(a0, b, c[0][j]);
            c[1][j] = wmma_bf16(a1, b, c[1][j]);
        }
    }
    #pragma unroll
    for (int i = 0; i < 2; ++i)
        #pragma unroll
        for (int j = 0; j < 4; ++j) {
            int n = N0 + j * 16 + mr;
            #pragma unroll
            for (int r = 0; r < 8; ++r) {
                int m = M0 + i * 16 + r + 8 * hf;
                size_t idx = (size_t)m * N + n;
                out[idx] += alpha[(m >> 11) * 1024 + n] * (c[i][j][r] + bias[n]);
            }
        }
}

// ---------------------------------------------------------------------------
// Host launcher
// ---------------------------------------------------------------------------
extern "C" void kernel_launch(void* const* d_in, const int* in_sizes, int n_in,
                              void* d_out, int out_size, void* d_ws, size_t ws_size,
                              hipStream_t stream) {
    (void)in_sizes; (void)n_in; (void)out_size; (void)ws_size;
    const float* x      = (const float*)d_in[0];
    const float* t      = (const float*)d_in[1];
    const float* agw    = (const float*)d_in[2];
    const float* abw    = (const float*)d_in[3];
    const float* Wq     = (const float*)d_in[4];
    const float* Wk     = (const float*)d_in[5];
    const float* Wv     = (const float*)d_in[6];
    const float* Wo     = (const float*)d_in[7];
    const float* aaw    = (const float*)d_in[8];
    const float* fgw    = (const float*)d_in[9];
    const float* fbw    = (const float*)d_in[10];
    const float* gatew  = (const float*)d_in[11];
    const float* hidw   = (const float*)d_in[12];
    const float* outw   = (const float*)d_in[13];
    const float* outb   = (const float*)d_in[14];
    const float* faw    = (const float*)d_in[15];
    float* out = (float*)d_out;

    const int M = 4 * 2048;   // 8192 tokens
    uint8_t* ws = (uint8_t*)d_ws;
    size_t off = 0;
    auto take = [&](size_t bytes) -> void* {
        void* p = ws + off;
        off += (bytes + 255) & ~(size_t)255;
        return p;
    };
    unsigned short* wq_bf = (unsigned short*)take((size_t)1024 * 1024 * 2);
    unsigned short* wk_bf = (unsigned short*)take((size_t)1024 * 1024 * 2);
    unsigned short* wv_bf = (unsigned short*)take((size_t)1024 * 1024 * 2);
    unsigned short* wo_bf = (unsigned short*)take((size_t)1024 * 1024 * 2);
    unsigned short* wg_bf = (unsigned short*)take((size_t)4096 * 1024 * 2);
    unsigned short* wh_bf = (unsigned short*)take((size_t)4096 * 1024 * 2);
    unsigned short* ww_bf = (unsigned short*)take((size_t)1024 * 4096 * 2);
    float* a_gamma = (float*)take(4 * 1024 * 4);
    float* a_beta  = (float*)take(4 * 1024 * 4);
    float* a_alpha = (float*)take(4 * 1024 * 4);
    float* f_gamma = (float*)take(4 * 1024 * 4);
    float* f_beta  = (float*)take(4 * 1024 * 4);
    float* f_alpha = (float*)take(4 * 1024 * 4);
    unsigned short* xn  = (unsigned short*)take((size_t)M * 1024 * 2);
    unsigned short* qb  = (unsigned short*)take((size_t)M * 1024 * 2);
    unsigned short* kb  = (unsigned short*)take((size_t)M * 1024 * 2);
    unsigned short* vb  = (unsigned short*)take((size_t)M * 1024 * 2);
    unsigned short* ob  = (unsigned short*)take((size_t)M * 1024 * 2);
    unsigned short* hb  = (unsigned short*)take((size_t)M * 4096 * 2);

    auto cvt = [&](const float* s, unsigned short* dptr, int n) {
        cvt_bf16<<<dim3(n / 1024), dim3(256), 0, stream>>>(s, dptr, n);
    };
    cvt(Wq, wq_bf, 1024 * 1024);
    cvt(Wk, wk_bf, 1024 * 1024);
    cvt(Wv, wv_bf, 1024 * 1024);
    cvt(Wo, wo_bf, 1024 * 1024);
    cvt(gatew, wg_bf, 4096 * 1024);
    cvt(hidw,  wh_bf, 4096 * 1024);
    cvt(outw,  ww_bf, 1024 * 4096);

    tvec<<<dim3(4, 4), 256, 0, stream>>>(t, agw, a_gamma);
    tvec<<<dim3(4, 4), 256, 0, stream>>>(t, abw, a_beta);
    tvec<<<dim3(4, 4), 256, 0, stream>>>(t, aaw, a_alpha);
    tvec<<<dim3(4, 4), 256, 0, stream>>>(t, fgw, f_gamma);
    tvec<<<dim3(4, 4), 256, 0, stream>>>(t, fbw, f_beta);
    tvec<<<dim3(4, 4), 256, 0, stream>>>(t, faw, f_alpha);

    // attn branch
    ada_rmsnorm_bf16<<<dim3(M / 8), 256, 0, stream>>>(x, a_gamma, a_beta, xn);
    gemm_proj<<<dim3(M / 64, 1024 / 256), 256, 0, stream>>>(xn, wq_bf, qb, 1024, 1024);
    gemm_proj<<<dim3(M / 64, 1024 / 256), 256, 0, stream>>>(xn, wk_bf, kb, 1024, 1024);
    gemm_proj<<<dim3(M / 64, 1024 / 256), 256, 0, stream>>>(xn, wv_bf, vb, 1024, 1024);
    flash_attn<<<dim3(16, 16, 4), 256, 0, stream>>>(qb, kb, vb, ob);
    gemm_wo_res<<<dim3(M / 64, 1024 / 256), 256, 0, stream>>>(ob, wo_bf, x, a_alpha, out);

    // ffn branch (out currently holds x1)
    ada_rmsnorm_bf16<<<dim3(M / 8), 256, 0, stream>>>(out, f_gamma, f_beta, xn);
    gemm_swiglu<<<dim3(M / 64, 4096 / 128), 256, 0, stream>>>(xn, wg_bf, wh_bf, hb);
    gemm_out_res<<<dim3(M / 64, 1024 / 256), 256, 0, stream>>>(hb, ww_bf, outb, f_alpha, out);
}
